// FusionBlock_76527727280509
// MI455X (gfx1250) — matrix operand order repaired and checked by reading, not compiled, not run on hardware
//
#include <hip/hip_runtime.h>

// ---------------------------------------------------------------------------
// CDNA5 (gfx1250) fusion block: channel-attn gate -> window attention +
// local convs -> sep conv -> MLP.  Dense GEMMs on v_wmma_f32_16x16x32_bf16
// (wave32, f32 accumulate), A-tiles staged by the Tensor Data Mover (TDM)
// with hardware LDS padding, B-tiles staged transposed so fragment reads are
// ds_load_b128 instead of scalar ds_load_u16 chains.
// ---------------------------------------------------------------------------

typedef __attribute__((ext_vector_type(16))) __bf16 v16bf;
typedef __attribute__((ext_vector_type(8)))  float  v8f;
typedef __attribute__((ext_vector_type(4)))  unsigned int u32x4;
typedef __attribute__((ext_vector_type(8)))  int    i32x8;
typedef __attribute__((ext_vector_type(4)))  int    i32x4;

#if __has_builtin(__builtin_amdgcn_tensor_load_to_lds) && \
    __has_builtin(__builtin_amdgcn_s_wait_tensorcnt)
#define USE_TDM 1
#else
#define USE_TDM 0
#endif

constexpr int Bn  = 4;
constexpr int Cc  = 256;
constexpr int Hh  = 128;
constexpr int Wd  = 128;
constexpr int HWp = Hh * Wd;      // 16384 = GEMM N per batch
constexpr int NT  = HWp;

#if USE_TDM
// 2D TDM tile load: rows x rowElems (bf16), global row stride strideD0 elems,
// LDS destination padded 4 DWORDs every 16 DWORDs (64B rows -> stride 40 elems)
__device__ __forceinline__ void tdm_load_2d(const __bf16* gaddr, unsigned ldsOff,
                                            int rowElems, int rows,
                                            int tensorD0, int tensorD1, int strideD0)
{
  const unsigned long long ga = (unsigned long long)(size_t)gaddr;
  u32x4 g0 = {0u, 0u, 0u, 0u};
  g0[0] = 1u;                                            // count=1 (valid), user mode
  g0[1] = ldsOff;                                        // lds_addr
  g0[2] = (unsigned)(ga & 0xFFFFFFFFull);                // global_addr[31:0]
  g0[3] = (unsigned)((ga >> 32) & 0x01FFFFFFull) | (2u << 30);  // addr[56:32] | type=2
  i32x8 g1 = {0, 0, 0, 0, 0, 0, 0, 0};
  g1[0] = (1 << 16) | (1 << 20) | (3 << 22) | (3 << 25); // data_size=2B, pad 4dw/16dw
  g1[1] = (tensorD0 & 0xFFFF) << 16;                     // tensor_dim0[15:0]
  g1[2] = ((tensorD0 >> 16) & 0xFFFF) | ((tensorD1 & 0xFFFF) << 16);
  g1[3] = ((tensorD1 >> 16) & 0xFFFF) | ((rowElems & 0xFFFF) << 16);   // tile_dim0
  g1[4] = (rows & 0xFFFF);                               // tile_dim1 (tile_dim2=0)
  g1[5] = strideD0;                                      // tensor_dim0_stride[31:0]
  i32x4 g2 = {0, 0, 0, 0};
  i32x4 g3 = {0, 0, 0, 0};
#if __clang_major__ >= 23
  i32x8 g4 = {0, 0, 0, 0, 0, 0, 0, 0};
  __builtin_amdgcn_tensor_load_to_lds(g0, g1, g2, g3, g4, 0);
#else
  __builtin_amdgcn_tensor_load_to_lds(g0, g1, g2, g3, 0);
#endif
}
#endif

// ---------------------------------------------------------------------------
// Generic tiled WMMA GEMM:  out[b] = A(MxK) * Bm[b](KxN) (+bias,+relu6,+resid)
// Block tile 64x64, K-step 32, 4 waves, each wave 32x32 (2x2 WMMA tiles).
// ---------------------------------------------------------------------------
__global__ __launch_bounds__(128)
void k_gemm(const __bf16* __restrict__ A, const __bf16* __restrict__ Bm,
            int M, int K,
            const float* __restrict__ bias, int doRelu6,
            const float* __restrict__ resid,
            float* __restrict__ outF,
            __bf16* __restrict__ outB,
            const float* __restrict__ obS, const float* __restrict__ obB)
{
  __shared__ __align__(16) __bf16 As[64][40];   // 64 rows x 32 (pad->40)
  __shared__ __align__(16) __bf16 Bt[64][40];   // B tile TRANSPOSED: [col][k]

  const int n0 = blockIdx.x * 64;
  const int m0 = blockIdx.y * 64;
  const int z  = blockIdx.z;
  const size_t bOffB = (size_t)z * K * NT;
  const size_t bOffO = (size_t)z * M * NT;

  const int t    = threadIdx.x;
  const int lane = t & 31;
  const int wv   = t >> 5;
  const int wr   = (wv >> 1) * 32;      // wave tile row
  const int wc   = (wv & 1)  * 32;      // wave tile col
  const int lrow = lane & 15;
  const int hi   = lane >> 4;

  v8f acc[2][2];
  #pragma unroll
  for (int i = 0; i < 2; i++)
    #pragma unroll
    for (int j = 0; j < 2; j++)
      #pragma unroll
      for (int r = 0; r < 8; r++) acc[i][j][r] = 0.f;

  const int arow = t >> 1, aseg = (t & 1) * 16;   // A: 64 rows, 2 thr/row
  const int brow = t >> 2, bseg = (t & 3) * 16;   // B: 32 rows, 4 thr/row

#if USE_TDM
  const unsigned ldsAs = (unsigned)(size_t)&As[0][0];
#endif

  for (int k0 = 0; k0 < K; k0 += 32) {
#if USE_TDM
    // ---- A tile via Tensor Data Mover (wave 0 issues; HW applies LDS pad)
    if (t < 32) {
      tdm_load_2d(A + (size_t)m0 * K + k0, ldsAs, 32, 64, K, M, K);
    }
#else
    {
      const uint4* ga = reinterpret_cast<const uint4*>(
          A + (size_t)(m0 + arow) * K + k0 + aseg);
      uint4 a0 = ga[0], a1 = ga[1];
      *reinterpret_cast<uint4*>(&As[arow][aseg])     = a0;
      *reinterpret_cast<uint4*>(&As[arow][aseg + 8]) = a1;
    }
#endif
    // ---- B tile: vector global load, transposed scalar stores into LDS
    {
      const __bf16* gb = Bm + bOffB + (size_t)(k0 + brow) * NT + n0 + bseg;
      __bf16 tmp[16];
      *reinterpret_cast<uint4*>(&tmp[0]) = *reinterpret_cast<const uint4*>(gb);
      *reinterpret_cast<uint4*>(&tmp[8]) = *reinterpret_cast<const uint4*>(gb + 8);
      #pragma unroll
      for (int i = 0; i < 16; i++) Bt[bseg + i][brow] = tmp[i];

      if (k0 + 32 < K) {   // prefetch next K tile -> global_prefetch_b8
        __builtin_prefetch(A + (size_t)(m0 + arow) * K + k0 + 32 + aseg, 0, 1);
        __builtin_prefetch(gb + 32, 0, 1);
      }
    }
#if USE_TDM
    if (t < 32) __builtin_amdgcn_s_wait_tensorcnt(0);
#endif
    __syncthreads();

    // ---- fragments per ISA 7.12.2 layouts (all reads contiguous -> b128)
    v16bf af[2], bfr[2];
    #pragma unroll
    for (int tr = 0; tr < 2; tr++) {
      const int ar = wr + tr * 16 + lrow;
      #pragma unroll
      for (int i = 0; i < 8; i++) {
        af[tr][i]     = As[ar][hi * 8 + i];        // K 0..7  / 8..15
        af[tr][i + 8] = As[ar][16 + hi * 8 + i];   // K 16..23 / 24..31
      }
    }
    #pragma unroll
    for (int tc = 0; tc < 2; tc++) {
      const int col = wc + tc * 16 + lrow;
      #pragma unroll
      for (int i = 0; i < 16; i++) bfr[tc][i] = Bt[col][(i < 8) ? (hi * 8 + i) : (16 + hi * 8 + (i - 8))];
    }

    #pragma unroll
    for (int tr = 0; tr < 2; tr++)
      #pragma unroll
      for (int tc = 0; tc < 2; tc++)
        acc[tr][tc] = __builtin_amdgcn_wmma_f32_16x16x32_bf16(
            false, af[tr], false, bfr[tc], (short)0, acc[tr][tc], false, false);
    __syncthreads();
  }

  // ---- epilogue (C/D layout: VGPR r -> M = hi*8 + r, N = lane&15)
  #pragma unroll
  for (int tr = 0; tr < 2; tr++)
    #pragma unroll
    for (int tc = 0; tc < 2; tc++) {
      const int gn  = n0 + wc + tc * 16 + lrow;
      const int gmb = m0 + wr + tr * 16 + hi * 8;
      #pragma unroll
      for (int r = 0; r < 8; r++) {
        const int gm = gmb + r;
        float v = acc[tr][tc][r];
        if (bias)    v += bias[gm];
        if (doRelu6) v = fminf(fmaxf(v, 0.f), 6.f);
        const size_t oi = bOffO + (size_t)gm * NT + gn;
        if (resid)   v += resid[oi];
        if (outF)    outF[oi] = v;
        if (outB) {
          float ov = obS ? (v * obS[gm] + obB[gm]) : v;
          outB[oi] = (__bf16)ov;
        }
      }
    }
}

// ---------------------------------------------------------------------------
// Implicit-GEMM 3x3 conv (local1, BN folded, 1x1 local2 folded into center
// tap by k_foldloc).  M=256, K=2304 = 256*9.  B tile is an im2col gather,
// stored transposed so fragment reads stay vectorized.
// ---------------------------------------------------------------------------
__global__ __launch_bounds__(128)
void k_gemm_loc(const __bf16* __restrict__ A, const __bf16* __restrict__ Y,
                const float* __restrict__ bias, __bf16* __restrict__ outB)
{
  constexpr int K = 2304;
  __shared__ __align__(16) __bf16 As[64][40];
  __shared__ __align__(16) __bf16 Bt[64][40];

  const int n0 = blockIdx.x * 64;
  const int m0 = blockIdx.y * 64;
  const int z  = blockIdx.z;
  const __bf16* Yb = Y + ((size_t)z * Cc << 14);

  const int t    = threadIdx.x;
  const int lane = t & 31;
  const int wv   = t >> 5;
  const int wr   = (wv >> 1) * 32;
  const int wc   = (wv & 1)  * 32;
  const int lrow = lane & 15;
  const int hi   = lane >> 4;

  v8f acc[2][2];
  #pragma unroll
  for (int i = 0; i < 2; i++)
    #pragma unroll
    for (int j = 0; j < 2; j++)
      #pragma unroll
      for (int r = 0; r < 8; r++) acc[i][j][r] = 0.f;

  const int arow = t >> 1, aseg = (t & 1) * 16;
  const int kk   = t & 31, cseg = (t >> 5) * 16;   // B: gather 16 cols per thr

#if USE_TDM
  const unsigned ldsAs = (unsigned)(size_t)&As[0][0];
#endif

  for (int k0 = 0; k0 < K; k0 += 32) {
#if USE_TDM
    if (t < 32) {
      tdm_load_2d(A + (size_t)m0 * K + k0, ldsAs, 32, 64, K, Cc, K);
    }
#else
    {
      const uint4* ga = reinterpret_cast<const uint4*>(
          A + (size_t)(m0 + arow) * K + k0 + aseg);
      uint4 a0 = ga[0], a1 = ga[1];
      *reinterpret_cast<uint4*>(&As[arow][aseg])     = a0;
      *reinterpret_cast<uint4*>(&As[arow][aseg + 8]) = a1;
    }
#endif
    // implicit im2col gather: kglob -> (cin, dy, dx); col -> (h, w)
    {
      const int kg = k0 + kk;
      const int ci = kg / 9;
      const int t9 = kg - ci * 9;
      const int dy = t9 / 3 - 1;
      const int dx = t9 - (t9 / 3) * 3 - 1;
      const __bf16* plane = Yb + ((size_t)ci << 14);
      #pragma unroll
      for (int i = 0; i < 16; i++) {
        const int n  = n0 + cseg + i;
        const int ih = (n >> 7) + dy;
        const int iw = (n & 127) + dx;
        __bf16 v = (__bf16)0.0f;
        if ((unsigned)ih < 128u && (unsigned)iw < 128u)
          v = plane[(ih << 7) + iw];
        Bt[cseg + i][kk] = v;          // transposed store
      }
    }
#if USE_TDM
    if (t < 32) __builtin_amdgcn_s_wait_tensorcnt(0);
#endif
    __syncthreads();

    v16bf af[2], bfr[2];
    #pragma unroll
    for (int tr = 0; tr < 2; tr++) {
      const int ar = wr + tr * 16 + lrow;
      #pragma unroll
      for (int i = 0; i < 8; i++) {
        af[tr][i]     = As[ar][hi * 8 + i];
        af[tr][i + 8] = As[ar][16 + hi * 8 + i];
      }
    }
    #pragma unroll
    for (int tc = 0; tc < 2; tc++) {
      const int col = wc + tc * 16 + lrow;
      #pragma unroll
      for (int i = 0; i < 16; i++) bfr[tc][i] = Bt[col][(i < 8) ? (hi * 8 + i) : (16 + hi * 8 + (i - 8))];
    }
    #pragma unroll
    for (int tr = 0; tr < 2; tr++)
      #pragma unroll
      for (int tc = 0; tc < 2; tc++)
        acc[tr][tc] = __builtin_amdgcn_wmma_f32_16x16x32_bf16(
            false, af[tr], false, bfr[tc], (short)0, acc[tr][tc], false, false);
    __syncthreads();
  }

  const size_t bOffO = (size_t)z * Cc * NT;
  #pragma unroll
  for (int tr = 0; tr < 2; tr++)
    #pragma unroll
    for (int tc = 0; tc < 2; tc++) {
      const int gn  = n0 + wc + tc * 16 + lrow;
      const int gmb = m0 + wr + tr * 16 + hi * 8;
      #pragma unroll
      for (int r = 0; r < 8; r++) {
        const int gm = gmb + r;
        float v = acc[tr][tc][r] + bias[gm];
        outB[bOffO + (size_t)gm * NT + gn] = (__bf16)v;
      }
    }
}

// ---------------------------------------------------------------------------
// Weight prep
// ---------------------------------------------------------------------------
__global__ void k_cvt(const float* __restrict__ src, __bf16* __restrict__ dst, int n)
{
  int gid = blockIdx.x * 256 + threadIdx.x;
  if (gid < n) dst[gid] = (__bf16)src[gid];
}

__global__ void k_foldloc(const float* __restrict__ l1w, const float* __restrict__ l1s,
                          const float* __restrict__ l1b, const float* __restrict__ l2w,
                          const float* __restrict__ l2s, const float* __restrict__ l2b,
                          __bf16* __restrict__ w, float* __restrict__ bias)
{
  int gid = blockIdx.x * 256 + threadIdx.x;
  if (gid < 256) bias[gid] = l1b[gid] + l2b[gid];
  if (gid >= 256 * 2304) return;
  const int o    = gid / 2304;
  const int rest = gid - o * 2304;
  const int ci   = rest / 9;
  const int t9   = rest - ci * 9;
  float v = l1s[o] * l1w[gid];                       // bn scale folded in
  if (t9 == 4) v += l2s[o] * l2w[o * 256 + ci];      // 1x1 -> center tap
  w[gid] = (__bf16)v;
}

// ---------------------------------------------------------------------------
// Channel attention: pooled stats (avg / max / softpool) then tiny MLP gate.
// ---------------------------------------------------------------------------
__global__ __launch_bounds__(256)
void k_pool(const float* __restrict__ x, const float* __restrict__ sc_,
            const float* __restrict__ bi_,
            float* __restrict__ avg, float* __restrict__ mx, float* __restrict__ soft)
{
  __shared__ float r0[256], r1[256];
  const int c = blockIdx.x, z = blockIdx.y;
  const float* p = x + (((size_t)z * 256 + c) << 14);
  const float sc = sc_[c], bi = bi_[c];
  float ls = 0.f, lm = -1e30f;
  for (int i = threadIdx.x; i < 16384; i += 256) {
    float v = fmaf(p[i], sc, bi);
    ls += v; lm = fmaxf(lm, v);
  }
  r0[threadIdx.x] = ls; r1[threadIdx.x] = lm;
  __syncthreads();
  for (int s = 128; s > 0; s >>= 1) {
    if (threadIdx.x < s) {
      r0[threadIdx.x] += r0[threadIdx.x + s];
      r1[threadIdx.x] = fmaxf(r1[threadIdx.x], r1[threadIdx.x + s]);
    }
    __syncthreads();
  }
  const float gsum = r0[0], gmax = r1[0];
  __syncthreads();
  float le = 0.f, lex = 0.f;
  for (int i = threadIdx.x; i < 16384; i += 256) {
    float v = fmaf(p[i], sc, bi);
    float e = __expf(v - gmax);
    le += e; lex += e * v;
  }
  r0[threadIdx.x] = le; r1[threadIdx.x] = lex;
  __syncthreads();
  for (int s = 128; s > 0; s >>= 1) {
    if (threadIdx.x < s) {
      r0[threadIdx.x] += r0[threadIdx.x + s];
      r1[threadIdx.x] += r1[threadIdx.x + s];
    }
    __syncthreads();
  }
  if (threadIdx.x == 0) {
    const int idx = z * 256 + c;
    avg[idx]  = gsum * (1.f / 16384.f);
    mx[idx]   = gmax;
    soft[idx] = r1[0] / r0[0];
  }
}

__global__ __launch_bounds__(256)
void k_camlp(const float* __restrict__ avg, const float* __restrict__ mx,
             const float* __restrict__ soft,
             const float* __restrict__ w1, const float* __restrict__ b1,
             const float* __restrict__ w2, const float* __restrict__ b2,
             float* __restrict__ sOut)
{
  __shared__ float hid[3][128];
  const int z = blockIdx.x;
  const float* pools[3] = {avg + z * 256, mx + z * 256, soft + z * 256};
  if (threadIdx.x < 128) {
    const int j = threadIdx.x;
    const float* wr = w1 + j * 256;
    for (int p = 0; p < 3; p++) {
      float s = b1[j];
      const float* pp = pools[p];
      for (int c = 0; c < 256; c++) s += wr[c] * pp[c];
      hid[p][j] = fmaxf(s, 0.f);
    }
  }
  __syncthreads();
  const int c = threadIdx.x;
  float acc = 3.f * b2[c];
  const float* wr = w2 + c * 128;
  for (int p = 0; p < 3; p++)
    for (int j = 0; j < 128; j++) acc += wr[j] * hid[p][j];
  sOut[z * 256 + c] = 1.f / (1.f + __expf(-acc));
}

// bn affine (+ optional per-(b,c) gate) -> bf16
__global__ void k_bnbf(const float* __restrict__ src, const float* __restrict__ s_,
                       const float* __restrict__ b_, const float* __restrict__ gate,
                       __bf16* __restrict__ dst, int n)
{
  int gid = blockIdx.x * 256 + threadIdx.x;
  if (gid >= n) return;
  const int c = (gid >> 14) & 255;
  float v = fmaf(src[gid], s_[c], b_[c]);
  if (gate) v *= gate[(gid >> 22) * 256 + c];
  dst[gid] = (__bf16)v;
}

// ---------------------------------------------------------------------------
// Window attention: one block per (window, head, batch); 64 threads = 64 rows.
// ---------------------------------------------------------------------------
__global__ __launch_bounds__(64)
void k_attn(const __bf16* __restrict__ qkv, const float* __restrict__ relb,
            __bf16* __restrict__ o)
{
  __shared__ float qs[64][17], ks[64][17], vs[64][17];
  const int win  = blockIdx.x;
  const int head = blockIdx.y;
  const int z    = blockIdx.z;
  const int wy = win >> 4, wx = win & 15;
  const int i  = threadIdx.x;
  const int hw = (((wy << 3) + (i >> 3)) << 7) + (wx << 3) + (i & 7);
  const __bf16* base = qkv + (size_t)z * 768 * HWp;
  const int ch = head * 16;
  for (int d = 0; d < 16; d++) {
    qs[i][d] = (float)base[(size_t)(      ch + d) * HWp + hw];
    ks[i][d] = (float)base[(size_t)(256 + ch + d) * HWp + hw];
    vs[i][d] = (float)base[(size_t)(512 + ch + d) * HWp + hw];
  }
  __syncthreads();

  float qr[16];
  #pragma unroll
  for (int d = 0; d < 16; d++) qr[d] = qs[i][d];

  float dr[64];
  const int yi = i >> 3, xi = i & 7;
  float m = -1e30f;
  for (int j = 0; j < 64; j++) {
    float s = 0.f;
    #pragma unroll
    for (int d = 0; d < 16; d++) s += qr[d] * ks[j][d];
    const int idx = (yi - (j >> 3) + 7) * 15 + (xi - (j & 7) + 7);
    s = s * 0.25f + relb[idx * 16 + head];    // d^-0.5 = 0.25
    dr[j] = s; m = fmaxf(m, s);
  }
  float se = 0.f;
  for (int j = 0; j < 64; j++) { float e = __expf(dr[j] - m); dr[j] = e; se += e; }
  const float inv = 1.f / se;
  float ov[16];
  #pragma unroll
  for (int d = 0; d < 16; d++) ov[d] = 0.f;
  for (int j = 0; j < 64; j++) {
    const float a = dr[j] * inv;
    #pragma unroll
    for (int d = 0; d < 16; d++) ov[d] += a * vs[j][d];
  }
  for (int d = 0; d < 16; d++)
    o[((size_t)z * 256 + ch + d) * HWp + hw] = (__bf16)ov[d];
}

// ax + ay (reflect+zero padded 8-tap averages, count_include_pad) + local
__global__ void k_axay(const __bf16* __restrict__ o, const __bf16* __restrict__ loc,
                       __bf16* __restrict__ outp, int n)
{
  int gid = blockIdx.x * 256 + threadIdx.x;
  if (gid >= n) return;
  const int w = gid & 127, h = (gid >> 7) & 127;
  const __bf16* pl = o + ((size_t)(gid >> 14) << 14);
  float ax = 0.f, ay = 0.f;
  #pragma unroll
  for (int t = -3; t <= 4; t++) {
    const int r = h + t;
    if (r >= 0 && r <= 128) { const int rr = (r == 128) ? 126 : r; ax += (float)pl[(rr << 7) + w]; }
    const int cw = w + t;
    if (cw >= 0 && cw <= 128) { const int cc = (cw == 128) ? 126 : cw; ay += (float)pl[(h << 7) + cc]; }
  }
  const float v = (ax + ay) * 0.125f + (float)loc[gid];
  outp[gid] = (__bf16)v;
}

// depthwise 8x8 (reflect-pad-1 then zero-pad-3) + sep BN -> bf16
__global__ __launch_bounds__(256)
void k_dw(const __bf16* __restrict__ src, const float* __restrict__ wgt,
          const float* __restrict__ ss, const float* __restrict__ sb,
          __bf16* __restrict__ dst)
{
  __shared__ float wl[64];
  const int c = blockIdx.y, z = blockIdx.z;
  if (threadIdx.x < 64) wl[threadIdx.x] = wgt[c * 64 + threadIdx.x];
  __syncthreads();
  const int pix = blockIdx.x * 256 + threadIdx.x;
  const int h = pix >> 7, w = pix & 127;
  const __bf16* pl = src + (((size_t)z * 256 + c) << 14);
  float s = 0.f;
  for (int r = 0; r < 8; r++) {
    int ih = h + r - 3;
    if (ih < 0 || ih > 128) continue;
    if (ih == 128) ih = 126;
    for (int q = 0; q < 8; q++) {
      const int iw = w + q - 3;
      if (iw < 0 || iw > 128) continue;
      const int iw2 = (iw == 128) ? 126 : iw;
      s += wl[r * 8 + q] * (float)pl[(ih << 7) + iw2];
    }
  }
  s = fmaf(s, ss[c], sb[c]);
  dst[(((size_t)z * 256 + c) << 14) + pix] = (__bf16)s;
}

// ---------------------------------------------------------------------------
// Host launcher
// ---------------------------------------------------------------------------
extern "C" void kernel_launch(void* const* d_in, const int* in_sizes, int n_in,
                              void* d_out, int out_size, void* d_ws, size_t ws_size,
                              hipStream_t stream)
{
  (void)in_sizes; (void)n_in; (void)out_size; (void)ws_size;
  const float* x       = (const float*)d_in[0];
  const float* y       = (const float*)d_in[1];
  const float* normx_s = (const float*)d_in[2];
  const float* normx_b = (const float*)d_in[3];
  const float* normy_s = (const float*)d_in[4];
  const float* normy_b = (const float*)d_in[5];
  const float* ca_w1   = (const float*)d_in[6];
  const float* ca_b1   = (const float*)d_in[7];
  const float* ca_w2   = (const float*)d_in[8];
  const float* ca_b2   = (const float*)d_in[9];
  const float* qkv_w   = (const float*)d_in[10];
  const float* l1w     = (const float*)d_in[11];
  const float* l1s     = (const float*)d_in[12];
  const float* l1b     = (const float*)d_in[13];
  const float* l2w     = (const float*)d_in[14];
  const float* l2s     = (const float*)d_in[15];
  const float* l2b     = (const float*)d_in[16];
  const float* sep_dw  = (const float*)d_in[17];
  const float* sep_s   = (const float*)d_in[18];
  const float* sep_b   = (const float*)d_in[19];
  const float* sep_pw  = (const float*)d_in[20];
  const float* relb    = (const float*)d_in[21];
  const float* mlp_w1  = (const float*)d_in[22];
  const float* mlp_b1  = (const float*)d_in[23];
  const float* mlp_w2  = (const float*)d_in[24];
  const float* mlp_b2  = (const float*)d_in[25];
  const float* norm2_s = (const float*)d_in[26];
  const float* norm2_b = (const float*)d_in[27];

  char* ws = (char*)d_ws;
  size_t off = 0;
  auto take = [&](size_t bytes) { size_t o = off; off += (bytes + 255) & ~(size_t)255; return o; };

  __bf16* wQkv = (__bf16*)(ws + take((size_t)768 * 256 * 2));
  __bf16* wLoc = (__bf16*)(ws + take((size_t)256 * 2304 * 2));
  float*  bLoc = (float*) (ws + take(256 * 4));
  __bf16* wPw  = (__bf16*)(ws + take((size_t)256 * 256 * 2));
  __bf16* wM1  = (__bf16*)(ws + take((size_t)1024 * 256 * 2));
  __bf16* wM2  = (__bf16*)(ws + take((size_t)256 * 1024 * 2));
  float*  stAvg  = (float*)(ws + take(1024 * 4));
  float*  stMax  = (float*)(ws + take(1024 * 4));
  float*  stSoft = (float*)(ws + take(1024 * 4));
  float*  stGate = (float*)(ws + take(1024 * 4));
  const size_t slabE = (size_t)Bn * Cc * HWp;            // 16.8M elems
  __bf16* slab2 = (__bf16*)(ws + take(slabE * 2));       // xn -> o -> bn2(x2)
  __bf16* slab3 = (__bf16*)(ws + take(slabE * 2));       // yn -> out_pre
  __bf16* slab4 = (__bf16*)(ws + take(slabE * 2));       // local -> dw
  __bf16* slab1 = (__bf16*)(ws + take((size_t)Bn * 768 * HWp * 2 * 2)); // qkv -> mlp hidden (4C)
  float*  x2    = (float*) (ws + take(slabE * 4));

  const int nEl = Bn * Cc * HWp;           // 16,777,216

  // 1) weights -> bf16 (BN folded for local path)
  k_cvt<<<(768 * 256 + 255) / 256, 256, 0, stream>>>(qkv_w, wQkv, 768 * 256);
  k_cvt<<<(256 * 256 + 255) / 256, 256, 0, stream>>>(sep_pw, wPw, 256 * 256);
  k_cvt<<<(1024 * 256 + 255) / 256, 256, 0, stream>>>(mlp_w1, wM1, 1024 * 256);
  k_cvt<<<(1024 * 256 + 255) / 256, 256, 0, stream>>>(mlp_w2, wM2, 1024 * 256);
  k_foldloc<<<(256 * 2304 + 255) / 256, 256, 0, stream>>>(l1w, l1s, l1b, l2w, l2s, l2b, wLoc, bLoc);

  // 2) channel-attention gate
  k_pool<<<dim3(256, 4), 256, 0, stream>>>(x, normx_s, normx_b, stAvg, stMax, stSoft);
  k_camlp<<<4, 256, 0, stream>>>(stAvg, stMax, stSoft, ca_w1, ca_b1, ca_w2, ca_b2, stGate);

  // 3) xn = bn(x)*gate, yn = bn(y)   (bf16)
  k_bnbf<<<nEl / 256, 256, 0, stream>>>(x, normx_s, normx_b, stGate, slab2, nEl);
  k_bnbf<<<nEl / 256, 256, 0, stream>>>(y, normy_s, normy_b, nullptr, slab3, nEl);

  // 4) qkv = Wqkv * xn  (768x256 GEMM, WMMA)
  k_gemm<<<dim3(256, 12, 4), 128, 0, stream>>>(wQkv, slab2, 768, 256,
      nullptr, 0, nullptr, nullptr, slab1, nullptr, nullptr);

  // 5) local = fused 3x3 implicit GEMM over yn (WMMA)
  k_gemm_loc<<<dim3(256, 4, 4), 128, 0, stream>>>(wLoc, slab3, bLoc, slab4);

  // 6) window attention -> o
  k_attn<<<dim3(256, 16, 4), 64, 0, stream>>>(slab1, relb, slab2);

  // 7) out_pre = ax + ay + local
  k_axay<<<nEl / 256, 256, 0, stream>>>(slab2, slab4, slab3, nEl);

  // 8) depthwise 8x8 + sep BN
  k_dw<<<dim3(64, 256, 4), 256, 0, stream>>>(slab3, sep_dw, sep_s, sep_b, slab4);

  // 9) pointwise: x2 = x + Wpw*dw ; also emit bn2(x2) bf16 for the MLP
  k_gemm<<<dim3(256, 4, 4), 128, 0, stream>>>(wPw, slab4, 256, 256,
      nullptr, 0, x, x2, slab2, norm2_s, norm2_b);

  // 10) MLP: h = relu6(W1*bn2 + b1) ; out = x2 + W2*h + b2
  k_gemm<<<dim3(256, 16, 4), 128, 0, stream>>>(wM1, slab2, 1024, 256,
      mlp_b1, 1, nullptr, nullptr, slab1, nullptr, nullptr);
  k_gemm<<<dim3(256, 4, 4), 128, 0, stream>>>(wM2, slab1, 256, 1024,
      mlp_b2, 0, x2, (float*)d_out, nullptr, nullptr, nullptr);
}
